// HomographyEstimator_75127567941823
// MI455X (gfx1250) — matrix operand order repaired and checked by reading, not compiled, not run on hardware
//
#include <hip/hip_runtime.h>

// CDNA5 (gfx1250) wave32 WMMA types
typedef __attribute__((ext_vector_type(16))) __bf16 v16bf;
typedef __attribute__((ext_vector_type(8)))  __bf16 v8bf;
typedef __attribute__((ext_vector_type(8)))  float  v8f;

// ---------------------------------------------------------------------------
// Zero-fill helper (ws is poisoned with 0xAA; we need a small zero region that
// OOB B-fragment loads are redirected to).
// ---------------------------------------------------------------------------
__global__ void zero_fill(__bf16* p, int n) {
    int i = blockIdx.x * blockDim.x + threadIdx.x;
    if (i < n) p[i] = (__bf16)0.0f;
}

// ---------------------------------------------------------------------------
// Weight repack: [Cout][Cin][5][5] fp32  ->  [Cout][25][Cinp] bf16 (zero pad
// ci >= Cin).  Makes A fragments two contiguous aligned b128 loads.
// ---------------------------------------------------------------------------
__global__ __launch_bounds__(256) void repack_w(const float* __restrict__ w,
                                                __bf16* __restrict__ wp,
                                                int Cin, int Cinp) {
    long idx = (long)blockIdx.x * blockDim.x + threadIdx.x;  // Cout*25*Cinp
    int  ci  = (int)(idx % Cinp);
    long t   = idx / Cinp;
    int  tap = (int)(t % 25);
    long m   = t / 25;
    float v = (ci < Cin) ? w[(m * Cin + ci) * 25 + tap] : 0.0f;
    wp[idx] = (__bf16)v;
}

// ---------------------------------------------------------------------------
// Grayscale: mean over 3 channels, fp32 -> bf16.  (B,3,128,128) -> (B,16384)
// ---------------------------------------------------------------------------
__global__ __launch_bounds__(256) void gray_mean(const float* __restrict__ in,
                                                 __bf16* __restrict__ out) {
    int idx = blockIdx.x * blockDim.x + threadIdx.x;  // 8*16384
    int b = idx >> 14;
    int p = idx & 16383;
    const float* base = in + (long)b * 3 * 16384 + p;
    float g = (base[0] + base[16384] + base[32768]) * (1.0f / 3.0f);
    out[idx] = (__bf16)g;
}

// ---------------------------------------------------------------------------
// Layer 0 direct conv (Cin=1 -> 64, 128x128), NHWC output. Only 0.8 GFLOP
// total; WMMA would waste 7/32 of K on padding, VALU is fine here.
// ---------------------------------------------------------------------------
__global__ __launch_bounds__(256) void conv0_direct(const __bf16* __restrict__ gs,
                                                    const float* __restrict__ w,
                                                    const float* __restrict__ bias,
                                                    __bf16* __restrict__ out) {
    int idx = blockIdx.x * blockDim.x + threadIdx.x;  // 8*128*128*64
    int co = idx & 63;
    int p  = idx >> 6;
    int x  = p & 127;
    int y  = (p >> 7) & 127;
    int b  = p >> 14;
    const __bf16* g = gs + (long)b * 16384;
    const float*  wr = w + co * 25;
    float acc = bias[co];
#pragma unroll
    for (int ty = 0; ty < 5; ++ty) {
        int iy = y + ty - 2;
        if ((unsigned)iy < 128u) {
#pragma unroll
            for (int tx = 0; tx < 5; ++tx) {
                int ix = x + tx - 2;
                if ((unsigned)ix < 128u)
                    acc += (float)g[iy * 128 + ix] * wr[ty * 5 + tx];
            }
        }
    }
    acc = acc > 0.f ? acc : 0.f;
    out[idx] = (__bf16)acc;   // idx == ((b*128+y)*128+x)*64+co
}

// ---------------------------------------------------------------------------
// Generic 5x5 conv, pad=2, bias+ReLU, WMMA bf16 implicit GEMM. NHWC.
//   in : bf16 [B][H][W][Cin]   (Cin % 32 == 0)
//   wp : bf16 [Cout][25][Cin]  (repacked)
//   out: bf16 [B][H][W][Cout]  (Cout % 32 == 0, (H*W) % 32 == 0, W % 16 == 0)
// One wave computes a 2x2 block of 16x16 WMMA tiles: 32 Cout x 32 spatial.
// 4 accumulators; each A fragment is shared by 2 B tiles and vice versa
// (16 FLOP/B of wave traffic instead of 8).  Inner loop is branch-free:
// OOB taps redirect the B load base to a zeroed buffer via per-lane pointer
// selects (stride forced to 0), so EXEC stays all-ones for every WMMA.
// ---------------------------------------------------------------------------
__global__ __launch_bounds__(256) void conv5x5_wmma(const __bf16* __restrict__ in,
                                                    const __bf16* __restrict__ wp,
                                                    const float* __restrict__ bias,
                                                    __bf16* __restrict__ out,
                                                    const __bf16* __restrict__ zbuf,
                                                    int Cin, int Cout, int H, int W) {
    const int wave = (blockIdx.x * blockDim.x + threadIdx.x) >> 5;
    const int lane = threadIdx.x & 31;
    const int l16  = lane & 15;
    const int lh   = lane >> 4;

    const int HW      = H * W;
    const int tiles_n = HW >> 5;    // 32-wide spatial blocks
    const int tiles_m = Cout >> 5;  // 32-wide Cout blocks
    const int tpb     = tiles_m * tiles_n;

    const int b     = wave / tpb;
    const int rem   = wave - b * tpb;
    const int mt    = rem / tiles_n;
    const int nt    = rem - mt * tiles_n;
    const int mbase = mt << 5;
    const int nbase = nt << 5;

    // Two spatial sub-tiles (B-fragment columns)
    const int n0 = nbase + l16;
    const int n1 = n0 + 16;
    const int x0 = n0 % W, y0 = n0 / W;
    const int x1 = n1 % W, y1 = n1 / W;

    const __bf16* __restrict__ inb = in + (long)b * HW * Cin;
    // Two A-fragment rows (Cout sub-tiles)
    const long aoff0 = (long)(mbase + l16) * 25;
    const long aoff1 = (long)(mbase + 16 + l16) * 25;
    const int  csteps = Cin >> 5;

    v8f acc00 = {0.f,0.f,0.f,0.f,0.f,0.f,0.f,0.f};
    v8f acc01 = acc00, acc10 = acc00, acc11 = acc00;

    int tap = 0;
    for (int ty = 0; ty < 5; ++ty) {
        const int iy0 = y0 + ty - 2;
        const int iy1 = y1 + ty - 2;
        for (int tx = 0; tx < 5; ++tx, ++tap) {
            const int ix0 = x0 + tx - 2;
            const int ix1 = x1 + tx - 2;
            const bool o0 = ((unsigned)iy0 >= (unsigned)H) | ((unsigned)ix0 >= (unsigned)W);
            const bool o1 = ((unsigned)iy1 >= (unsigned)H) | ((unsigned)ix1 >= (unsigned)W);

            const __bf16* br0 = inb + ((long)(o0 ? 0 : iy0) * W + (o0 ? 0 : ix0)) * Cin + (lh << 4);
            const __bf16* br1 = inb + ((long)(o1 ? 0 : iy1) * W + (o1 ? 0 : ix1)) * Cin + (lh << 4);
            const __bf16* bb0 = o0 ? zbuf : br0;
            const __bf16* bb1 = o1 ? zbuf : br1;
            const int     s0  = o0 ? 0 : 32;
            const int     s1  = o1 ? 0 : 32;

            const __bf16* ar0 = wp + (aoff0 + tap) * Cin + (lh << 4);
            const __bf16* ar1 = wp + (aoff1 + tap) * Cin + (lh << 4);

            for (int cb = 0; cb < csteps; ++cb) {
                v16bf bf0 = *(const v16bf*)(bb0 + cb * s0);     // 32B aligned
                v16bf bf1 = *(const v16bf*)(bb1 + cb * s1);

                const __bf16* a0p = ar0 + (cb << 5);
                v8bf a0lo = *(const v8bf*)a0p;                  // 16B aligned
                v8bf a0hi = *(const v8bf*)(a0p + 8);
                v16bf af0 = __builtin_shufflevector(a0lo, a0hi,
                    0,1,2,3,4,5,6,7,8,9,10,11,12,13,14,15);

                const __bf16* a1p = ar1 + (cb << 5);
                v8bf a1lo = *(const v8bf*)a1p;
                v8bf a1hi = *(const v8bf*)(a1p + 8);
                v16bf af1 = __builtin_shufflevector(a1lo, a1hi,
                    0,1,2,3,4,5,6,7,8,9,10,11,12,13,14,15);

                acc00 = __builtin_amdgcn_wmma_f32_16x16x32_bf16(
                    false, af0, false, bf0, (short)0, acc00, false, false);
                acc01 = __builtin_amdgcn_wmma_f32_16x16x32_bf16(
                    false, af0, false, bf1, (short)0, acc01, false, false);
                acc10 = __builtin_amdgcn_wmma_f32_16x16x32_bf16(
                    false, af1, false, bf0, (short)0, acc10, false, false);
                acc11 = __builtin_amdgcn_wmma_f32_16x16x32_bf16(
                    false, af1, false, bf1, (short)0, acc11, false, false);
            }
        }
    }

    // Epilogue: bias + ReLU; D layout M = msub + 8*lh + r, N = nsub + l16.
    const v8f* accs[4] = {&acc00, &acc01, &acc10, &acc11};
#pragma unroll
    for (int mi = 0; mi < 2; ++mi) {
#pragma unroll
        for (int ni = 0; ni < 2; ++ni) {
            const v8f& a = *accs[mi * 2 + ni];
            const int mb = mbase + (mi << 4) + (lh << 3);
            const int nn = (ni ? n1 : n0);
            v8bf res;
#pragma unroll
            for (int r = 0; r < 8; ++r) {
                float v = a[r] + bias[mb + r];
                v = v > 0.f ? v : 0.f;
                res[r] = (__bf16)v;
            }
            *(v8bf*)(out + ((long)b * HW + nn) * Cout + mb) = res;
        }
    }
}

// ---------------------------------------------------------------------------
// 2x2 max pool stride 2, NHWC bf16. [8][H][H][C] -> [8][H/2][H/2][C]
// ---------------------------------------------------------------------------
__global__ __launch_bounds__(256) void maxpool2(const __bf16* __restrict__ in,
                                                __bf16* __restrict__ out,
                                                int C, int H) {
    int idx = blockIdx.x * blockDim.x + threadIdx.x;  // 8*(H/2)*(H/2)*C
    int W = H, ow = W >> 1;
    int c = idx % C;
    int p = idx / C;
    int ox = p % ow;
    int t  = p / ow;
    int oy = t % ow;
    int b  = t / ow;
    const __bf16* src = in + (((long)(b * H + oy * 2)) * W + ox * 2) * C + c;
    float v0 = (float)src[0],           v1 = (float)src[C];
    float v2 = (float)src[(long)W * C], v3 = (float)src[(long)W * C + C];
    float m = fmaxf(fmaxf(v0, v1), fmaxf(v2, v3));
    out[idx] = (__bf16)m;
}

// ---------------------------------------------------------------------------
// L2 normalize over C=128, NHWC at 16x16. grid=8 (batch), block=256 (ij).
// ---------------------------------------------------------------------------
__global__ __launch_bounds__(256) void l2norm_c(const __bf16* __restrict__ in,
                                                __bf16* __restrict__ out) {
    int b = blockIdx.x;
    int ij = threadIdx.x;
    const __bf16* p = in + ((long)b * 256 + ij) * 128;
    float s = 0.f;
#pragma unroll 8
    for (int c = 0; c < 128; ++c) {
        float v = (float)p[c];
        s += v * v;
    }
    float inv = 1.0f / fmaxf(sqrtf(s), 1e-12f);
    __bf16* q = out + ((long)b * 256 + ij) * 128;
#pragma unroll 8
    for (int c = 0; c < 128; ++c)
        q[c] = (__bf16)((float)p[c] * inv);
}

// ---------------------------------------------------------------------------
// Cost volume, NHWC output with channel pad to 1120:
// cv[b][i][j][d] = leaky( dot_c(fa[b,i,j,:], fb[b,i+dy-16,j+dx-16,:]) / 128 ),
// d = dy*33+dx for d<1089, zero for the 31 pad channels.
// ---------------------------------------------------------------------------
__global__ __launch_bounds__(256) void cost_volume(const __bf16* __restrict__ fa,
                                                   const __bf16* __restrict__ fb,
                                                   __bf16* __restrict__ cv) {
    int idx = blockIdx.x * blockDim.x + threadIdx.x;  // 8*256*1120
    int d  = idx % 1120;
    int t  = idx / 1120;
    int ij = t & 255;
    int b  = t >> 8;
    float r = 0.f;
    if (d < 1089) {
        int i = ij >> 4, j = ij & 15;
        int dy = d / 33 - 16;
        int dx = d % 33 - 16;
        int pi = i + dy, pj = j + dx;
        float acc = 0.f;
        if ((unsigned)pi < 16u && (unsigned)pj < 16u) {
            const __bf16* pa = fa + ((long)b * 256 + ij) * 128;
            const __bf16* pb = fb + ((long)b * 256 + pi * 16 + pj) * 128;
#pragma unroll 8
            for (int c = 0; c < 128; ++c)
                acc += (float)pa[c] * (float)pb[c];
        }
        acc *= (1.0f / 128.0f);
        r = acc > 0.f ? acc : 0.1f * acc;
    }
    cv[idx] = (__bf16)r;
}

// ---------------------------------------------------------------------------
// FC1: 131072 -> 1024 + ReLU, batch 8. Bandwidth-bound on 536MB fp32 weights
// (read exactly once); act buffer is NHWC so remap k_ref = c*256+ij.
// ---------------------------------------------------------------------------
__global__ __launch_bounds__(256) void fc1_kernel(const __bf16* __restrict__ in,
                                                  const float* __restrict__ w,
                                                  const float* __restrict__ bias,
                                                  float* __restrict__ out) {
    int o = blockIdx.x;
    int t = threadIdx.x;
    const float* wr = w + (long)o * 131072;
    float acc[8] = {0.f, 0.f, 0.f, 0.f, 0.f, 0.f, 0.f, 0.f};
    for (int k = t; k < 131072; k += 256) {
        __builtin_prefetch(&wr[k + 4096], 0, 1);   // global_prefetch_b8
        float wv = wr[k];
        int c  = k >> 8;
        int ij = k & 255;
        int ao = (ij << 9) + c;                    // NHWC offset of k_ref
#pragma unroll
        for (int b = 0; b < 8; ++b)
            acc[b] += wv * (float)in[(long)b * 131072 + ao];
    }
    __shared__ float red[256];
    for (int b = 0; b < 8; ++b) {
        red[t] = acc[b];
        __syncthreads();
        for (int off = 128; off > 0; off >>= 1) {
            if (t < off) red[t] += red[t + off];
            __syncthreads();
        }
        if (t == 0) {
            float v = red[0] + bias[o];
            out[b * 1024 + o] = v > 0.f ? v : 0.f;
        }
        __syncthreads();
    }
}

// ---------------------------------------------------------------------------
// FC2: 1024 -> 8, batch 8.
// ---------------------------------------------------------------------------
__global__ void fc2_kernel(const float* __restrict__ in,
                           const float* __restrict__ w,
                           const float* __restrict__ bias,
                           float* __restrict__ out) {
    int t = threadIdx.x;  // 0..63
    int b = t >> 3, o = t & 7;
    const float* wr = w + o * 1024;
    const float* ir = in + b * 1024;
    float acc = bias[o];
#pragma unroll 4
    for (int j = 0; j < 1024; ++j)
        acc += ir[j] * wr[j];
    out[t] = acc;
}

// ---------------------------------------------------------------------------
// Host orchestration
// ---------------------------------------------------------------------------
extern "C" void kernel_launch(void* const* d_in, const int* in_sizes, int n_in,
                              void* d_out, int out_size, void* d_ws, size_t ws_size,
                              hipStream_t stream) {
    const float* inA = (const float*)d_in[0];
    const float* inB = (const float*)d_in[1];

    const float *faW[8], *faB[8], *fbW[8], *fbB[8];
    const float *c1w, *c1b, *c2w, *c2b, *c3w, *c3b, *fc1w, *fc1b, *fc2w, *fc2b;

    if (n_in >= 44 && in_sizes[2] == 1600) {
        // insertion order: fa(8x w,b), fb(8x w,b), c1, c2, c3, fc1_w, fc1_b, fc2_w, fc2_b
        for (int i = 0; i < 8; ++i) {
            faW[i] = (const float*)d_in[2 + 2 * i];  faB[i] = (const float*)d_in[3 + 2 * i];
            fbW[i] = (const float*)d_in[18 + 2 * i]; fbB[i] = (const float*)d_in[19 + 2 * i];
        }
        c1w = (const float*)d_in[34]; c1b = (const float*)d_in[35];
        c2w = (const float*)d_in[36]; c2b = (const float*)d_in[37];
        c3w = (const float*)d_in[38]; c3b = (const float*)d_in[39];
        fc1w = (const float*)d_in[40]; fc1b = (const float*)d_in[41];
        fc2w = (const float*)d_in[42]; fc2b = (const float*)d_in[43];
    } else {
        // alphabetically sorted pytree: c1,c2,c3, fa, fb, fc1_b, fc1_w, fc2_b, fc2_w
        c1w = (const float*)d_in[2]; c1b = (const float*)d_in[3];
        c2w = (const float*)d_in[4]; c2b = (const float*)d_in[5];
        c3w = (const float*)d_in[6]; c3b = (const float*)d_in[7];
        for (int i = 0; i < 8; ++i) {
            faW[i] = (const float*)d_in[8 + 2 * i];  faB[i] = (const float*)d_in[9 + 2 * i];
            fbW[i] = (const float*)d_in[24 + 2 * i]; fbB[i] = (const float*)d_in[25 + 2 * i];
        }
        fc1b = (const float*)d_in[40]; fc1w = (const float*)d_in[41];
        fc2b = (const float*)d_in[42]; fc2w = (const float*)d_in[43];
    }

    // ---- workspace layout (bytes) ----
    char* ws = (char*)d_ws;
    __bf16* P0   = (__bf16*)(ws + 0);            // 16 MB ping
    __bf16* P1   = (__bf16*)(ws + 16777216);     // 16 MB pong
    __bf16* FN0  = (__bf16*)(ws + 33554432);     // 512 KB
    __bf16* FN1  = (__bf16*)(ws + 34078720);     // 512 KB
    __bf16* CVb  = (__bf16*)(ws + 34603008);     // 8*256*1120*2 = 4.59 MB
    __bf16* R0   = (__bf16*)(ws + 39190528);     // 2 MB
    __bf16* R1   = (__bf16*)(ws + 41287680);     // 2 MB
    float*  FC1O = (float*)(ws + 43384832);      // 32 KB
    __bf16* ZBUF = (__bf16*)(ws + 43417600);     // 256 B zeroed
    char*   wpBase = ws + 43417856;              // repacked weights (~63 MB)

    // extractor layer dims (layer 0 handled by conv0_direct)
    const int cinL[8]  = {1, 64, 64, 64, 64, 128, 128, 128};
    const int coutL[8] = {64, 64, 64, 64, 128, 128, 128, 128};

    __bf16* wpExt[2][8];
    {
        long off = 0;
        for (int s = 0; s < 2; ++s)
            for (int l = 1; l < 8; ++l) {
                wpExt[s][l] = (__bf16*)(wpBase + off);
                off += (long)coutL[l] * 25 * cinL[l] * 2;
            }
        // off = 6,963,200 here
    }
    __bf16* c1p = (__bf16*)(wpBase + 8172800);               // 512*25*1120*2
    __bf16* c2p = (__bf16*)(wpBase + 8172800 + 28672000);    // 512*25*512*2
    __bf16* c3p = (__bf16*)(wpBase + 8172800 + 28672000 + 13107200);

    // ---- prep: zero buffer + weight repacks ----
    zero_fill<<<1, 128, 0, stream>>>(ZBUF, 128);
    for (int s = 0; s < 2; ++s) {
        const float* const* Wt = s ? fbW : faW;
        for (int l = 1; l < 8; ++l) {
            long nel = (long)coutL[l] * 25 * cinL[l];
            repack_w<<<(int)(nel / 256), 256, 0, stream>>>(Wt[l], wpExt[s][l], cinL[l], cinL[l]);
        }
    }
    repack_w<<<(512L * 25 * 1120) / 256, 256, 0, stream>>>(c1w, c1p, 1089, 1120);
    repack_w<<<(512L * 25 * 512) / 256, 256, 0, stream>>>(c2w, c2p, 512, 512);
    repack_w<<<(512L * 25 * 512) / 256, 256, 0, stream>>>(c3w, c3p, 512, 512);

    auto conv = [&](const __bf16* in, const __bf16* wpk, const float* bs, __bf16* out,
                    int Cin, int Cout, int H) {
        int HW = H * H;
        int waves = 8 * (Cout >> 5) * (HW >> 5);  // 2x2 blocked tiles; multiple of 8
        conv5x5_wmma<<<waves / 8, 256, 0, stream>>>(in, wpk, bs, out, ZBUF, Cin, Cout, H, H);
    };
    auto pool = [&](const __bf16* in, __bf16* out, int C, int H) {
        int nel = 8 * C * (H >> 1) * (H >> 1);
        maxpool2<<<nel / 256, 256, 0, stream>>>(in, out, C, H);
    };

    __bf16* FN[2] = {FN0, FN1};
    for (int s = 0; s < 2; ++s) {
        const float* img = s ? inB : inA;
        const float* const* Bs = s ? fbB : faB;
        const float* w0 = (s ? fbW : faW)[0];

        gray_mean<<<131072 / 256, 256, 0, stream>>>(img, P0);
        conv0_direct<<<(8 * 16384 * 64) / 256, 256, 0, stream>>>(P0, w0, Bs[0], P1);
        conv(P1, wpExt[s][1], Bs[1], P0, 64, 64, 128);
        pool(P0, P1, 64, 128);                        // -> 64x64x64
        conv(P1, wpExt[s][2], Bs[2], P0, 64, 64, 64);
        conv(P0, wpExt[s][3], Bs[3], P1, 64, 64, 64);
        pool(P1, P0, 64, 64);                         // -> 32x32x64
        conv(P0, wpExt[s][4], Bs[4], P1, 64, 128, 32);
        conv(P1, wpExt[s][5], Bs[5], P0, 128, 128, 32);
        pool(P0, P1, 128, 32);                        // -> 16x16x128
        conv(P1, wpExt[s][6], Bs[6], P0, 128, 128, 16);
        conv(P0, wpExt[s][7], Bs[7], P1, 128, 128, 16);
        l2norm_c<<<8, 256, 0, stream>>>(P1, FN[s]);
    }

    cost_volume<<<(8 * 256 * 1120) / 256, 256, 0, stream>>>(FN0, FN1, CVb);

    conv(CVb, c1p, c1b, R0, 1120, 512, 16);
    conv(R0, c2p, c2b, R1, 512, 512, 16);
    conv(R1, c3p, c3b, R0, 512, 512, 16);

    fc1_kernel<<<1024, 256, 0, stream>>>(R0, fc1w, fc1b, FC1O);
    fc2_kernel<<<1, 64, 0, stream>>>(FC1O, fc2w, fc2b, (float*)d_out);

    (void)in_sizes; (void)n_in; (void)out_size; (void)ws_size;
}
